// BiLSTM_CRF_78941498901206
// MI455X (gfx1250) — matrix operand order repaired and checked by reading, not compiled, not run on hardware
//
#include <hip/hip_runtime.h>

// ---------------------------------------------------------------------------
// BiLSTM-CRF on MI455X (gfx1250, wave32).
// Roofline: ~43 GFLOP total. Parallel GEMMs -> bf16 WMMA (negligible time).
// Binding constraint = 4 sequential LSTM scans (2048 dependent steps) and the
// Viterbi scan. LSTM scan: W_hh (2MB bf16/dir) sharded across 8 WGPs' LDS
// (256KB each, loaded ONCE via Tensor Data Mover), so per-step traffic is
// only the 2KB hidden vector through L2 + one arrive/spin barrier.
// fwd+bwd directions run concurrently in one 16-WG launch.
// ---------------------------------------------------------------------------
#define L_SEQ   2048
#define E_DIM   512
#define H_DIM   512      // H = HID/2
#define HID_DIM 1024     // 2*H
#define G_DIM   2048     // 4*H
#define T_TAGS  64
#define NEG_VAL (-1000.0f)
#define NWG_DIR 8        // workgroups per direction in the LSTM scan
#define HSLICE  64       // hidden units owned per workgroup (512/8)

typedef __bf16        v16bf   __attribute__((ext_vector_type(16)));
typedef float         v8f     __attribute__((ext_vector_type(8)));
typedef unsigned int  u32x4   __attribute__((ext_vector_type(4)));
typedef int           i32x4   __attribute__((ext_vector_type(4)));
typedef int           i32x8   __attribute__((ext_vector_type(8)));

#ifndef __has_builtin
#define __has_builtin(x) 0
#endif
#if __has_builtin(__builtin_amdgcn_tensor_load_to_lds) && __has_builtin(__builtin_amdgcn_s_wait_tensorcnt)
#define USE_TDM 1
#else
#define USE_TDM 0
#endif

__device__ __forceinline__ unsigned short f32_to_bf16(float f) {
  unsigned int u = __float_as_uint(f);
  u += 0x7fffu + ((u >> 16) & 1u);          // round-to-nearest-even
  return (unsigned short)(u >> 16);
}
__device__ __forceinline__ float bf16lo(unsigned int u) { return __uint_as_float(u << 16); }
__device__ __forceinline__ float bf16hi(unsigned int u) { return __uint_as_float(u & 0xffff0000u); }
__device__ __forceinline__ float sigmoidf(float x) { return 1.0f / (1.0f + __expf(-x)); }

// ---------------------------------------------------------------------------
// f32 -> bf16 elementwise (weights converted once; live in L2 afterwards)
// ---------------------------------------------------------------------------
__global__ __launch_bounds__(256)
void cvt_bf16_kernel(const float* __restrict__ src, unsigned short* __restrict__ dst, size_t n) {
  size_t i = (size_t)blockIdx.x * blockDim.x + threadIdx.x;
  if (i < n) dst[i] = f32_to_bf16(src[i]);
}

// ---------------------------------------------------------------------------
// Embedding gather: x0[t,e] = bf16(emb[sentence[t], e])  (2048 of 50000 rows)
// ---------------------------------------------------------------------------
__global__ __launch_bounds__(256)
void gather_embed_kernel(const int* __restrict__ sent, const float* __restrict__ emb,
                         unsigned short* __restrict__ x0) {
  size_t i = (size_t)blockIdx.x * blockDim.x + threadIdx.x;
  if (i >= (size_t)L_SEQ * E_DIM) return;
  int t = (int)(i / E_DIM);
  int e = (int)(i % E_DIM);
  x0[i] = f32_to_bf16(emb[(size_t)sent[t] * E_DIM + e]);
}

// ---------------------------------------------------------------------------
// tiny init kernel for the scan barrier counters
// ---------------------------------------------------------------------------
__global__ void init_ctr_kernel(unsigned int* c) {
  if (threadIdx.x < 2) c[threadIdx.x] = 0u;
}

// ---------------------------------------------------------------------------
// WMMA bf16 GEMM:  C[M,N] = A[M,K] * B[N,K]^T + bias0[N] + bias1[N] (opt tanh)
// One wave per 32x32 output block (2x2 WMMA tiles): 8 b128 fragment loads and
// 4 v_wmma_f32_16x16x32_bf16 per K-step -> 2x FLOPs/byte vs one tile/wave.
// Fragment packing per CDNA5 ISA 7.12.2 (A and B alike: lane half h holds k 8h..8h+7, 16+8h..16+8h+7).
// ---------------------------------------------------------------------------
__global__ __launch_bounds__(32)
void wmma_gemm_bf16_kernel(const unsigned short* __restrict__ A,
                           const unsigned short* __restrict__ B,
                           float* __restrict__ C,
                           int M, int N, int K,
                           const float* __restrict__ bias0,
                           const float* __restrict__ bias1,
                           int applyTanh) {
  const int tileN = blockIdx.x * 32;
  const int tileM = blockIdx.y * 32;
  const int lane  = threadIdx.x;
  const int lo    = lane & 15;
  const int hi    = lane >> 4;

  const unsigned short* aRow0 = A + (size_t)(tileM + lo) * K;
  const unsigned short* aRow1 = aRow0 + (size_t)16 * K;
  const unsigned short* bCol0 = B + (size_t)(tileN + lo) * K;
  const unsigned short* bCol1 = bCol0 + (size_t)16 * K;

  union Frag { v16bf v; u32x4 q[2]; };
  v8f c00 = {}, c01 = {}, c10 = {}, c11 = {};

  for (int kk = 0; kk < K; kk += 32) {
    Frag a0, a1, b0, b1;
    a0.q[0] = *reinterpret_cast<const u32x4*>(aRow0 + kk + hi * 8);
    a0.q[1] = *reinterpret_cast<const u32x4*>(aRow0 + kk + 16 + hi * 8);
    a1.q[0] = *reinterpret_cast<const u32x4*>(aRow1 + kk + hi * 8);
    a1.q[1] = *reinterpret_cast<const u32x4*>(aRow1 + kk + 16 + hi * 8);
    b0.q[0] = *reinterpret_cast<const u32x4*>(bCol0 + kk + hi * 8);
    b0.q[1] = *reinterpret_cast<const u32x4*>(bCol0 + kk + 16 + hi * 8);
    b1.q[0] = *reinterpret_cast<const u32x4*>(bCol1 + kk + hi * 8);
    b1.q[1] = *reinterpret_cast<const u32x4*>(bCol1 + kk + 16 + hi * 8);
    if (kk + 32 < K) {
      __builtin_prefetch(aRow0 + kk + 32, 0, 0);  // global_prefetch_b8
      __builtin_prefetch(bCol0 + kk + 32, 0, 0);
    }
    c00 = __builtin_amdgcn_wmma_f32_16x16x32_bf16(false, a0.v, false, b0.v, (short)0, c00, false, false);
    c01 = __builtin_amdgcn_wmma_f32_16x16x32_bf16(false, a0.v, false, b1.v, (short)0, c01, false, false);
    c10 = __builtin_amdgcn_wmma_f32_16x16x32_bf16(false, a1.v, false, b0.v, (short)0, c10, false, false);
    c11 = __builtin_amdgcn_wmma_f32_16x16x32_bf16(false, a1.v, false, b1.v, (short)0, c11, false, false);
  }

  const int col0 = tileN + lo;
  const int col1 = tileN + 16 + lo;
  float bv0 = 0.0f, bv1 = 0.0f;
  if (bias0) { bv0 += bias0[col0]; bv1 += bias0[col1]; }
  if (bias1) { bv0 += bias1[col0]; bv1 += bias1[col1]; }
#pragma unroll
  for (int r = 0; r < 8; ++r) {
    const int row0 = tileM + r + hi * 8;
    const int row1 = tileM + 16 + r + hi * 8;
    float v00 = c00[r] + bv0, v01 = c01[r] + bv1;
    float v10 = c10[r] + bv0, v11 = c11[r] + bv1;
    if (applyTanh) { v00 = tanhf(v00); v01 = tanhf(v01); v10 = tanhf(v10); v11 = tanhf(v11); }
    C[(size_t)row0 * N + col0] = v00;
    C[(size_t)row0 * N + col1] = v01;
    C[(size_t)row1 * N + col0] = v10;
    C[(size_t)row1 * N + col1] = v11;
  }
}

// ---------------------------------------------------------------------------
// TDM: DMA this workgroup's W_hh shard (4 gate blocks x 64 rows x 512 bf16 =
// 256KB) global -> LDS in ONE tensor_load_to_lds. 3D tile: x=512 elems (row),
// y=64 rows (stride 512 elems), z=4 gate blocks (stride 512*512 elems).
// D# dword layout per CDNA5 ISA 08_async_tensor §8.3-8.5. This toolchain's
// builtin is the 6-arg form: (u32x4, i32x8, i32x4, i32x4, i32x8, cpol).
// ---------------------------------------------------------------------------
#if USE_TDM
__device__ __forceinline__ void tdm_load_whh_shard(unsigned lds_off, const void* gaddr) {
  unsigned long long ga = (unsigned long long)gaddr;
  u32x4 g0;
  g0[0] = 1u;                                          // count=1, user descriptor
  g0[1] = lds_off;                                     // lds_addr
  g0[2] = (unsigned)(ga & 0xffffffffu);                // global_addr[31:0]
  g0[3] = (unsigned)((ga >> 32) & 0x1ffffffu) | 0x80000000u;  // addr[56:32] | type=2

  i32x8 g1;
  g1[0] = (int)(1u << 16);           // workgroup_mask=0, data_size=1 (2 bytes)
  g1[1] = (int)(512u << 16);         // tensor_dim0[15:0]=512 (atomic_barrier_addr=0)
  g1[2] = (int)(64u << 16);          // tensor_dim0[31:16]=0 | tensor_dim1[15:0]=64
  g1[3] = (int)(512u << 16);         // tensor_dim1[31:16]=0 | tile_dim0=512
  g1[4] = (int)(64u | (4u << 16));   // tile_dim1=64 | tile_dim2=4
  g1[5] = (int)512u;                 // tensor_dim0_stride[31:0]=512
  g1[6] = 0;                         // stride0 hi | tensor_dim1_stride[15:0]=0
  g1[7] = (int)4u;                   // tensor_dim1_stride[47:16]: 262144>>16 = 4

  i32x4 g2;
  g2[0] = 4;                         // tensor_dim2 = 4
  g2[1] = 0;                         // tensor_dim3
  g2[2] = 0;                         // tensor_dim2_stride lo
  g2[3] = 0;                         // stride hi | tile_dim3=0
  i32x4 g3 = {0, 0, 0, 0};
  i32x8 g4 = {0, 0, 0, 0, 0, 0, 0, 0};

  __builtin_amdgcn_tensor_load_to_lds(g0, g1, g2, g3, g4, 0);
}
#endif

// ---------------------------------------------------------------------------
// Multi-workgroup persistent LSTM scan, BOTH directions in one launch.
// grid = 16 blocks x 256 threads: blocks 0-7 = forward, 8-15 = backward.
// Each block owns 64 hidden units -> 256 gate rows; its 256KB W_hh shard is
// LDS-resident for the whole scan. Per step:
//   g[own 256 rows] = xg[t] + W_shard @ h        (LDS reads only)
//   c,h update for own 64 units (c in registers), h slice -> global,
//   device-scope arrive/spin barrier, reload full 512-float h.
// h exchange is double-buffered by step parity (no read/write race).
// ---------------------------------------------------------------------------
__global__ __launch_bounds__(256)
void lstm_scan_kernel(const float* __restrict__ xg_f,        // [L,4H] fwd (biases folded)
                      const float* __restrict__ xg_b,        // [L,4H] bwd
                      const unsigned short* __restrict__ whh,// [2,4H,H] bf16
                      unsigned short* __restrict__ hout,     // [L,2H] bf16
                      float* __restrict__ hx,                // [2 dirs][2 slots][512]
                      unsigned int* __restrict__ ctrs) {     // [2]
  extern __shared__ char smem[];                             // 256KB weight shard
  __shared__ float h_lds[H_DIM];
  __shared__ float g_lds[4 * HSLICE];

  const int tid = threadIdx.x;
  const int dir = blockIdx.x >> 3;          // 0 fwd, 1 bwd
  const int wg  = blockIdx.x & (NWG_DIR - 1);
  const int n0  = wg * HSLICE;

  const float* xg = dir ? xg_b : xg_f;
  const unsigned short* whh_d = whh + (size_t)dir * G_DIM * H_DIM;
  float* hxd = hx + (size_t)dir * 2 * H_DIM;
  unsigned int* ctr = ctrs + dir;
  unsigned short* w_lds = (unsigned short*)smem;

  // ---- one-time weight shard preload: global -> LDS ----
  const unsigned short* gsrc = whh_d + (size_t)n0 * H_DIM;   // start of row n0, gate 0
#if USE_TDM
  if (tid < 32) {                                            // one wave issues the DMA
    tdm_load_whh_shard((unsigned)(size_t)w_lds, (const void*)gsrc);
    __builtin_amdgcn_s_wait_tensorcnt(0);
  }
#else
  // cooperative copy: 4 gate blocks of 64 contiguous rows (64KB each)
  for (int q = 0; q < 4; ++q) {
    const u32x4* src = reinterpret_cast<const u32x4*>(whh_d + ((size_t)q * H_DIM + n0) * H_DIM);
    u32x4* dst = reinterpret_cast<u32x4*>(w_lds + (size_t)q * HSLICE * H_DIM);
    for (int i = tid; i < HSLICE * H_DIM / 8; i += 256) dst[i] = src[i];
  }
#endif
  __syncthreads();

  const int q = tid >> 6;                   // gate index 0..3 (i,f,g,o)
  const int m = tid & 63;                   // unit within slice
  float c_reg = 0.0f;

  for (int s = 0; s < L_SEQ; ++s) {
    const int t = dir ? (L_SEQ - 1 - s) : s;

    // refresh full hidden vector (slot by parity; step 0 uses zeros)
    if (s == 0) {
      for (int k = tid; k < H_DIM; k += 256) h_lds[k] = 0.0f;
    } else {
      const float* hsrc = hxd + (size_t)(s & 1) * H_DIM;
      for (int k = tid; k < H_DIM; k += 256) h_lds[k] = hsrc[k];
    }
    __syncthreads();

    // gate row (q, n0+m): dot(W_shard_row, h) from LDS
    float acc = xg[(size_t)t * G_DIM + (size_t)q * H_DIM + n0 + m];
    const u32x4* wq = reinterpret_cast<const u32x4*>(w_lds + (size_t)tid * H_DIM);
#pragma unroll 4
    for (int k4 = 0; k4 < H_DIM / 8; ++k4) {
      const u32x4 u = wq[k4];
      const int kb = k4 * 8;
      acc += bf16lo(u.x) * h_lds[kb + 0] + bf16hi(u.x) * h_lds[kb + 1]
           + bf16lo(u.y) * h_lds[kb + 2] + bf16hi(u.y) * h_lds[kb + 3]
           + bf16lo(u.z) * h_lds[kb + 4] + bf16hi(u.z) * h_lds[kb + 5]
           + bf16lo(u.w) * h_lds[kb + 6] + bf16hi(u.w) * h_lds[kb + 7];
    }
    g_lds[tid] = acc;
    __syncthreads();

    // cell/hidden update for own 64 units
    if (tid < HSLICE) {
      const float ig = sigmoidf(g_lds[tid]);                 // i
      const float fg = sigmoidf(g_lds[HSLICE + tid]);        // f
      const float gg = tanhf(g_lds[2 * HSLICE + tid]);       // g
      const float og = sigmoidf(g_lds[3 * HSLICE + tid]);    // o
      c_reg = fg * c_reg + ig * gg;
      const float h = og * tanhf(c_reg);
      hxd[(size_t)((s + 1) & 1) * H_DIM + n0 + tid] = h;     // exchange slot
      hout[(size_t)t * HID_DIM + dir * H_DIM + n0 + tid] = f32_to_bf16(h);
    }
    __threadfence();
    __syncthreads();

    // device-scope arrive/spin barrier across this direction's 8 workgroups
    if (tid == 0) {
      atomicAdd(ctr, 1u);
      const unsigned target = (unsigned)NWG_DIR * (unsigned)(s + 1);
      while (__hip_atomic_load(ctr, __ATOMIC_ACQUIRE, __HIP_MEMORY_SCOPE_AGENT) < target) { }
    }
    __syncthreads();
  }
}

// ---------------------------------------------------------------------------
// Viterbi: single 64-thread workgroup; thread tt owns "next tag" tt.
// ---------------------------------------------------------------------------
__global__ __launch_bounds__(64)
void viterbi_kernel(const float* __restrict__ feats,   // [L, T]
                    const float* __restrict__ Amat,    // [T, T]
                    unsigned char* __restrict__ bptr,  // [L, T]
                    float* __restrict__ out) {         // [L+1]: path, score
  __shared__ float As[T_TAGS][T_TAGS + 1];
  __shared__ float delta[T_TAGS];
  __shared__ float term[T_TAGS];
  const int t = threadIdx.x;

  for (int i = t; i < T_TAGS * T_TAGS; i += T_TAGS)
    As[i / T_TAGS][i % T_TAGS] = Amat[i];
  delta[t] = (t == 0) ? 0.0f : NEG_VAL;
  __syncthreads();

  for (int s = 0; s < L_SEQ; ++s) {
    float best = delta[0] + As[t][0];
    int bi = 0;
#pragma unroll
    for (int p = 1; p < T_TAGS; ++p) {
      const float sc = delta[p] + As[t][p];
      if (sc > best) { best = sc; bi = p; }   // strict > == first-max argmax
    }
    const float nd = feats[(size_t)s * T_TAGS + t] + best;
    bptr[(size_t)s * T_TAGS + t] = (unsigned char)bi;
    __syncthreads();
    delta[t] = nd;
    __syncthreads();
  }

  term[t] = delta[t] + As[T_TAGS - 1][t];
  __syncthreads();

  if (t == 0) {
    int best = 0; float bs = term[0];
    for (int p = 1; p < T_TAGS; ++p)
      if (term[p] > bs) { bs = term[p]; best = p; }
    out[L_SEQ] = bs;
    out[L_SEQ - 1] = (float)best;
    int tag = best;
    for (int i = L_SEQ - 1; i >= 1; --i) {
      tag = (int)bptr[(size_t)i * T_TAGS + tag];
      out[i - 1] = (float)tag;
    }
  }
}

// ---------------------------------------------------------------------------
// Host orchestration
// ---------------------------------------------------------------------------
extern "C" void kernel_launch(void* const* d_in, const int* in_sizes, int n_in,
                              void* d_out, int out_size, void* d_ws, size_t ws_size,
                              hipStream_t stream) {
  (void)in_sizes; (void)n_in; (void)out_size; (void)ws_size;

  const int*   sent  = (const int*)  d_in[0];
  const float* emb   = (const float*)d_in[1];
  const float* w_ih0 = (const float*)d_in[2];
  const float* w_hh0 = (const float*)d_in[3];
  const float* b_ih0 = (const float*)d_in[4];
  const float* b_hh0 = (const float*)d_in[5];
  const float* w_ih1 = (const float*)d_in[6];
  const float* w_hh1 = (const float*)d_in[7];
  const float* b_ih1 = (const float*)d_in[8];
  const float* b_hh1 = (const float*)d_in[9];
  const float* W_tag = (const float*)d_in[10];
  const float* b_tag = (const float*)d_in[11];
  const float* Amat  = (const float*)d_in[12];

  const int L = L_SEQ, E = E_DIM, H = H_DIM, HID = HID_DIM, G = G_DIM, T = T_TAGS;

  // workspace carve-out (256B aligned), total ~63 MB
  char* ws = (char*)d_ws;
  size_t off = 0;
  auto carve = [&](size_t bytes) -> char* {
    char* p = ws + off;
    off += (bytes + 255) & ~(size_t)255;
    return p;
  };
  unsigned short* wih0_bf = (unsigned short*)carve((size_t)2 * G * E   * 2);
  unsigned short* whh0_bf = (unsigned short*)carve((size_t)2 * G * H   * 2);
  unsigned short* wih1_bf = (unsigned short*)carve((size_t)2 * G * HID * 2);
  unsigned short* whh1_bf = (unsigned short*)carve((size_t)2 * G * H   * 2);
  unsigned short* wtag_bf = (unsigned short*)carve((size_t)T * HID     * 2);
  unsigned short* x0      = (unsigned short*)carve((size_t)L * E       * 2);
  float*          xgf     = (float*)         carve((size_t)L * G       * 4);
  float*          xgb     = (float*)         carve((size_t)L * G       * 4);
  unsigned short* h1      = (unsigned short*)carve((size_t)L * HID     * 2);
  unsigned short* h2      = (unsigned short*)carve((size_t)L * HID     * 2);
  float*          feats   = (float*)         carve((size_t)L * T       * 4);
  unsigned char*  bp      = (unsigned char*) carve((size_t)L * T);
  float*          hx      = (float*)         carve((size_t)2 * 2 * H   * 4);
  unsigned int*   ctrs    = (unsigned int*)  carve(256);

  auto cvt = [&](const float* s, unsigned short* d, size_t n) {
    cvt_bf16_kernel<<<(unsigned)((n + 255) / 256), 256, 0, stream>>>(s, d, n);
  };
  cvt(w_ih0, wih0_bf, (size_t)2 * G * E);
  cvt(w_hh0, whh0_bf, (size_t)2 * G * H);
  cvt(w_ih1, wih1_bf, (size_t)2 * G * HID);
  cvt(w_hh1, whh1_bf, (size_t)2 * G * H);
  cvt(W_tag, wtag_bf, (size_t)T * HID);

  {
    size_t n = (size_t)L * E;
    gather_embed_kernel<<<(unsigned)((n + 255) / 256), 256, 0, stream>>>(sent, emb, x0);
  }

  auto gemm = [&](const unsigned short* A, const unsigned short* B, float* C,
                  int M, int N, int K, const float* bb0, const float* bb1, int tnh) {
    dim3 grid(N / 32, M / 32);
    wmma_gemm_bf16_kernel<<<grid, 32, 0, stream>>>(A, B, C, M, N, K, bb0, bb1, tnh);
  };

  const size_t lstm_lds = (size_t)4 * HSLICE * H_DIM * 2;   // 256 KB dynamic LDS

  // ---- layer 0 ----
  gemm(x0, wih0_bf,                 xgf, L, G, E, b_ih0,     b_hh0,     0);
  gemm(x0, wih0_bf + (size_t)G * E, xgb, L, G, E, b_ih0 + G, b_hh0 + G, 0);
  init_ctr_kernel<<<1, 32, 0, stream>>>(ctrs);
  lstm_scan_kernel<<<2 * NWG_DIR, 256, lstm_lds, stream>>>(xgf, xgb, whh0_bf, h1, hx, ctrs);

  // ---- layer 1 ----
  gemm(h1, wih1_bf,                   xgf, L, G, HID, b_ih1,     b_hh1,     0);
  gemm(h1, wih1_bf + (size_t)G * HID, xgb, L, G, HID, b_ih1 + G, b_hh1 + G, 0);
  init_ctr_kernel<<<1, 32, 0, stream>>>(ctrs);
  lstm_scan_kernel<<<2 * NWG_DIR, 256, lstm_lds, stream>>>(xgf, xgb, whh1_bf, h2, hx, ctrs);

  // ---- tag projection: feats = tanh(h2 @ W_tag^T + b_tag) ----
  gemm(h2, wtag_bf, feats, L, T, HID, b_tag, nullptr, 1);

  // ---- Viterbi decode ----
  viterbi_kernel<<<1, T, 0, stream>>>(feats, Amat, bp, (float*)d_out);
}
